// UMambaEncoder_34462817583165
// MI455X (gfx1250) — matrix operand order repaired and checked by reading, not compiled, not run on hardware
//
#include <hip/hip_runtime.h>
#include <hip/hip_bf16.h>
#include <math.h>

// ---------------------------------------------------------------------------
// CDNA5 (gfx1250) UMamba encoder forward.
// Conv1d + all projections run as implicit-GEMM bf16 WMMA kernels
// (v_wmma_f32_16x16x32_bf16, wave32, fragments per ISA 7.12.2 layouts),
// software-pipelined: global loads of tile k+1 overlap WMMAs of tile k.
// LDS staging uses packed bf16 pairs (ds_store_b32) and b64 global loads.
// ---------------------------------------------------------------------------

typedef __attribute__((ext_vector_type(16))) __bf16 v16bf;
typedef __attribute__((ext_vector_type(8)))  float  v8f;

union FragU { uint4 u[2]; v16bf v; };

#define MTILE 64
#define NTILE 128
#define KTILE 32
#define KPAD  40   // 32 + 8 pad halves -> 80B rows (multiple of 16B for b128 LDS loads)

__device__ __forceinline__ float softplus_f(float x) {
  return (x > 20.f) ? x : log1pf(expf(x));
}
__device__ __forceinline__ float silu_f(float x) {
  return x * (1.f / (1.f + expf(-x)));
}
__device__ __forceinline__ unsigned pack_bf16(float a, float b) {
  union { __bf16 h[2]; unsigned u; } t;
  t.h[0] = (__bf16)a; t.h[1] = (__bf16)b;
  return t.u;
}

// ---------------- implicit-GEMM conv1d (NCH), bf16 WMMA --------------------
// out[b][co][l] = sum_{ci,kw} w[co][ci][kw] * x[b][ci][l*stride + kw - pad]
// GEMM view: M=Cout, N=B*Lout, K=Cin*Kw.  Requires Lout % NTILE == 0
// (true for this net: Lout in {16384,8192,4096,2048,1024}).
__global__ __launch_bounds__(256) void conv1d_wmma(
    const float* __restrict__ xin, const float* __restrict__ wgt,
    float* __restrict__ out,
    int Cin, int Lin, int Cout, int Lout, int Kw, int pad, int stride)
{
  __shared__ __align__(16) __bf16 As[MTILE][KPAD];   // [m][k]
  __shared__ __align__(16) __bf16 Bs[NTILE][KPAD];   // [n][k]
  const int tid  = threadIdx.x;
  const int lane = tid & 31;
  const int wave = tid >> 5;
  const int m0 = blockIdx.y * MTILE;
  const int n0 = blockIdx.x * NTILE;
  const int Ktot = Cin * Kw;
  const int m_off = (wave >> 1) * 16;   // 4 M sub-tiles
  const int n_off = (wave & 1) * 64;    // 2 N sub-tiles of 64 (4 accs of 16)

  // whole 128-wide N tile sits inside one batch element
  const int bblk  = n0 / Lout;
  const int lbase = n0 - bblk * Lout;
  const float* xb = xin + (size_t)bblk * Cin * Lin;

  const int kp = (tid & 15) * 2;  // this thread's k-pair within tile (invariant)
  const int r0 = tid >> 4;        // staging row base 0..15 (invariant)

  float aReg[8];    // 4 rows x 2 k
  float bReg[16];   // 8 rows x 2 k

  auto ldA = [&](int k0) {
    int k = k0 + kp;
    bool ok0 = (k < Ktot), ok1 = (k + 1 < Ktot);
#pragma unroll
    for (int j = 0; j < 4; ++j) { // w[m][k] (k = ci*Kw+kw) is flat m*Ktot+k
      const float* p = wgt + (size_t)(m0 + r0 + j * 16) * Ktot + k;
      aReg[2 * j]     = ok0 ? p[0] : 0.f;
      aReg[2 * j + 1] = ok1 ? p[1] : 0.f;
    }
  };
  auto ldB = [&](int k0) {
    int k = k0 + kp;
    int ci0 = k / Kw, kw0 = k - ci0 * Kw;            // one division per K-step
    int ci1 = ci0, kw1 = kw0 + 1;
    if (kw1 == Kw) { kw1 = 0; ++ci1; }
    bool ok0 = (k < Ktot), ok1 = (k + 1 < Ktot);
    const float* c0 = xb + (size_t)ci0 * Lin;
    const float* c1 = xb + (size_t)ci1 * Lin;
    int off0 = kw0 - pad, off1 = kw1 - pad;
#pragma unroll
    for (int j = 0; j < 8; ++j) {
      int ls = (lbase + r0 + j * 16) * stride;
      int l0 = ls + off0, l1 = ls + off1;
      bReg[2 * j]     = (ok0 && l0 >= 0 && l0 < Lin) ? c0[l0] : 0.f;
      bReg[2 * j + 1] = (ok1 && l1 >= 0 && l1 < Lin) ? c1[l1] : 0.f;
    }
  };
  auto stTile = [&]() {
#pragma unroll
    for (int j = 0; j < 4; ++j)
      *(unsigned*)&As[r0 + j * 16][kp] = pack_bf16(aReg[2 * j], aReg[2 * j + 1]);
#pragma unroll
    for (int j = 0; j < 8; ++j)
      *(unsigned*)&Bs[r0 + j * 16][kp] = pack_bf16(bReg[2 * j], bReg[2 * j + 1]);
  };

  v8f acc[4] = {};
  ldA(0); ldB(0);

  for (int k0 = 0; k0 < Ktot; k0 += KTILE) {
    __syncthreads();                 // previous tile's fragments consumed
    stTile();
    __syncthreads();
    if (k0 + KTILE < Ktot) { ldA(k0 + KTILE); ldB(k0 + KTILE); }  // overlap

    // A fragment: 16x32 bf16 (lane<16: K 0-7 & 16-23; lane>=16: +8)
    const int arow = m_off + (lane & 15);
    const int kh = (lane & 16) ? 8 : 0;
    FragU af;
    af.u[0] = *(const uint4*)&As[arow][kh];
    af.u[1] = *(const uint4*)&As[arow][16 + kh];
    // B fragments: column = lane&15 (lane<16: K 0-7 & 16-23; lane>=16: +8)
    const int kb = (lane & 16) ? 8 : 0;
    const int col = lane & 15;
    FragU bfr[4];
#pragma unroll
    for (int t = 0; t < 4; ++t) {
      bfr[t].u[0] = *(const uint4*)&Bs[n_off + t * 16 + col][kb];
      bfr[t].u[1] = *(const uint4*)&Bs[n_off + t * 16 + col][kb + 16];
    }
#pragma unroll
    for (int t = 0; t < 4; ++t)
      acc[t] = __builtin_amdgcn_wmma_f32_16x16x32_bf16(
          false, af.v, false, bfr[t].v, (short)0, acc[t], false, false);
  }

  // D layout: VGPR i -> row (lane<16 ? i : i+8), col lane&15
  float* ob = out + (size_t)bblk * Cout * Lout;
#pragma unroll
  for (int t = 0; t < 4; ++t) {
    int l = lbase + n_off + t * 16 + (lane & 15);
#pragma unroll
    for (int i = 0; i < 8; ++i) {
      int m = m0 + m_off + ((lane & 16) ? i + 8 : i);
      ob[(size_t)m * Lout + l] = acc[t][i];
    }
  }
}

// ---------------- generic row-major GEMM: out[n][m] = act[n][:].w[m][:] ----
// act: (Nrows, K) with row stride lda; wgt: (M, K) row-major.
// K and lda must be even (true here: K in {8,32,128,256,512,1024}).
// layout 0: out[n*ldo + m].  layout 1: out[((n/Lseq)*M + m)*Lseq + n%Lseq]
// (layout 1 requires Lseq % NTILE == 0; true here).
// actmode: 0 none, 1 softplus.  bias: per-m, nullable.
__global__ __launch_bounds__(256) void gemm_wmma(
    const float* __restrict__ act, const float* __restrict__ wgt,
    float* __restrict__ out,
    int Nrows, int M, int K, int lda, int ldo,
    const float* __restrict__ bias, int actmode, int layout, int Lseq)
{
  __shared__ __align__(16) __bf16 As[MTILE][KPAD];
  __shared__ __align__(16) __bf16 Bs[NTILE][KPAD];
  const int tid  = threadIdx.x;
  const int lane = tid & 31;
  const int wave = tid >> 5;
  const int m0 = blockIdx.y * MTILE;
  const int n0 = blockIdx.x * NTILE;
  const int m_off = (wave >> 1) * 16;
  const int n_off = (wave & 1) * 64;

  const int kp = (tid & 15) * 2;
  const int r0 = tid >> 4;

  float2 aReg[4];
  float2 bReg[8];

  auto ldA = [&](int k0) {
    int k = k0 + kp;
    bool ok = (k < K);      // K even -> k+1 < K as well
#pragma unroll
    for (int j = 0; j < 4; ++j) {
      int mg = m0 + r0 + j * 16;
      aReg[j] = (ok && mg < M) ? *(const float2*)&wgt[(size_t)mg * K + k]
                               : float2{0.f, 0.f};
    }
  };
  auto ldB = [&](int k0) {
    int k = k0 + kp;
    bool ok = (k < K);
#pragma unroll
    for (int j = 0; j < 8; ++j)
      bReg[j] = ok ? *(const float2*)&act[(size_t)(n0 + r0 + j * 16) * lda + k]
                   : float2{0.f, 0.f};
  };
  auto stTile = [&]() {
#pragma unroll
    for (int j = 0; j < 4; ++j)
      *(unsigned*)&As[r0 + j * 16][kp] = pack_bf16(aReg[j].x, aReg[j].y);
#pragma unroll
    for (int j = 0; j < 8; ++j)
      *(unsigned*)&Bs[r0 + j * 16][kp] = pack_bf16(bReg[j].x, bReg[j].y);
  };

  v8f acc[4] = {};
  ldA(0); ldB(0);

  for (int k0 = 0; k0 < K; k0 += KTILE) {
    __syncthreads();
    stTile();
    __syncthreads();
    if (k0 + KTILE < K) { ldA(k0 + KTILE); ldB(k0 + KTILE); }

    const int arow = m_off + (lane & 15);
    const int kh = (lane & 16) ? 8 : 0;
    FragU af;
    af.u[0] = *(const uint4*)&As[arow][kh];
    af.u[1] = *(const uint4*)&As[arow][16 + kh];
    const int kb = (lane & 16) ? 8 : 0;
    const int col = lane & 15;
    FragU bfr[4];
#pragma unroll
    for (int t = 0; t < 4; ++t) {
      bfr[t].u[0] = *(const uint4*)&Bs[n_off + t * 16 + col][kb];
      bfr[t].u[1] = *(const uint4*)&Bs[n_off + t * 16 + col][kb + 16];
    }
#pragma unroll
    for (int t = 0; t < 4; ++t)
      acc[t] = __builtin_amdgcn_wmma_f32_16x16x32_bf16(
          false, af.v, false, bfr[t].v, (short)0, acc[t], false, false);
  }

  const int bblk  = (layout == 1) ? (n0 / Lseq) : 0;
  const int lbase = (layout == 1) ? (n0 - bblk * Lseq) : 0;
#pragma unroll
  for (int t = 0; t < 4; ++t) {
    int nloc = n_off + t * 16 + (lane & 15);
#pragma unroll
    for (int i = 0; i < 8; ++i) {
      int m = m0 + m_off + ((lane & 16) ? i + 8 : i);
      if (m < M) {
        float v = acc[t][i];
        if (bias) v += bias[m];
        if (actmode == 1) v = softplus_f(v);
        if (layout == 0) {
          out[(size_t)(n0 + nloc) * ldo + m] = v;
        } else {
          out[((size_t)bblk * M + m) * Lseq + (lbase + nloc)] = v;
        }
      }
    }
  }
}

// ---------------- BatchNorm statistics: mean/var per channel over (B,L) ----
__global__ __launch_bounds__(256) void bn_stats(
    const float* __restrict__ x, float* __restrict__ mean, float* __restrict__ var,
    int B, int C, int L)
{
  const int c = blockIdx.x;
  __shared__ float s1[256];
  __shared__ float s2[256];
  const int tid = threadIdx.x;
  float a = 0.f, q = 0.f;
  for (int b = 0; b < B; ++b) {
    const float* p = x + ((size_t)b * C + c) * L;
    for (int l = tid; l < L; l += 256) {
      float v = p[l];
      a += v; q += v * v;
    }
  }
  s1[tid] = a; s2[tid] = q;
  __syncthreads();
  for (int st = 128; st > 0; st >>= 1) {
    if (tid < st) { s1[tid] += s1[tid + st]; s2[tid] += s2[tid + st]; }
    __syncthreads();
  }
  if (tid == 0) {
    float inv = 1.f / (float)(B * L);
    float m = s1[0] * inv;
    mean[c] = m;
    var[c] = s2[0] * inv - m * m;
  }
}

// ---------------- BN normalize + optional residual + LeakyReLU -------------
__global__ __launch_bounds__(256) void bn_act(
    const float* __restrict__ in, const float* __restrict__ res,
    const float* __restrict__ mean, const float* __restrict__ var,
    const float* __restrict__ g, const float* __restrict__ bb,
    float* __restrict__ out, int C, int L, size_t total)
{
  size_t idx = (size_t)blockIdx.x * blockDim.x + threadIdx.x;
  if (idx >= total) return;
  int c = (int)((idx / (size_t)L) % (size_t)C);
  float v = (in[idx] - mean[c]) * rsqrtf(var[c] + 1e-5f) * g[c] + bb[c];
  if (res) v += res[idx];
  out[idx] = (v >= 0.f) ? v : 0.01f * v;
}

// ---------------- LayerNorm over channel dim, (B,C,L) -> (B*L, C) ----------
__global__ __launch_bounds__(128) void ln_rows(
    const float* __restrict__ x, const float* __restrict__ g,
    const float* __restrict__ bb, float* __restrict__ out, int C, int L)
{
  const int row = blockIdx.x;          // row = b*L + l
  const int b = row / L, l = row - b * L;
  __shared__ float s1[128];
  __shared__ float s2[128];
  __shared__ float mu, rstd;
  const int tid = threadIdx.x;
  float a = 0.f, q = 0.f;
  for (int c = tid; c < C; c += 128) {
    float v = x[((size_t)b * C + c) * L + l];
    a += v; q += v * v;
  }
  s1[tid] = a; s2[tid] = q;
  __syncthreads();
  for (int st = 64; st > 0; st >>= 1) {
    if (tid < st) { s1[tid] += s1[tid + st]; s2[tid] += s2[tid + st]; }
    __syncthreads();
  }
  if (tid == 0) {
    float m = s1[0] / (float)C;
    mu = m;
    rstd = rsqrtf(s2[0] / (float)C - m * m + 1e-5f);
  }
  __syncthreads();
  for (int c = tid; c < C; c += 128) {
    float v = x[((size_t)b * C + c) * L + l];
    out[(size_t)row * C + c] = (v - mu) * rstd * g[c] + bb[c];
  }
}

// ---------------- depthwise causal conv (D_CONV=4) + bias + SiLU -----------
// xz: (B*L, 2*di) rows (in_proj output), reads columns [0, di).
// xc out: (B*L, di).
__global__ __launch_bounds__(256) void dwconv_silu(
    const float* __restrict__ xz, const float* __restrict__ cw,
    const float* __restrict__ cb, float* __restrict__ xc,
    int L, int di, size_t total)
{
  size_t idx = (size_t)blockIdx.x * blockDim.x + threadIdx.x;
  if (idx >= total) return;
  int d = (int)(idx % (size_t)di);
  size_t bl = idx / (size_t)di;
  int l = (int)(bl % (size_t)L);
  size_t brow = bl - (size_t)l;        // b*L
  float s = cb[d];
#pragma unroll
  for (int j = 0; j < 4; ++j) {
    int ll = l + j - 3;                // left-pad 3 (causal)
    if (ll >= 0) s += cw[d * 4 + j] * xz[(brow + ll) * (size_t)(2 * di) + d];
  }
  xc[idx] = silu_f(s);
}

// ---------------- selective scan (sequential over L), fused D & SiLU gate --
// One thread per (b, d); 16 SSM states held in registers.
__global__ __launch_bounds__(256) void mamba_scan(
    const float* __restrict__ dt, const float* __restrict__ xc,
    const float* __restrict__ xz, const float* __restrict__ dbl,
    const float* __restrict__ A_log, const float* __restrict__ Dp,
    float* __restrict__ y, int Bdi, int L, int di, int r)
{
  int idx = blockIdx.x * blockDim.x + threadIdx.x;
  if (idx >= Bdi) return;
  int b = idx / di, d = idx - b * di;
  const int R2N = r + 32;
  float A[16];
#pragma unroll
  for (int n = 0; n < 16; ++n) A[n] = -expf(A_log[(size_t)d * 16 + n]);
  const float Dv = Dp[d];
  float h[16];
#pragma unroll
  for (int n = 0; n < 16; ++n) h[n] = 0.f;

  const float* pdt = dt + (size_t)b * L * di + d;
  const float* pxc = xc + (size_t)b * L * di + d;
  const float* pz  = xz + (size_t)b * L * (2 * di) + di + d;
  const float* pbc = dbl + (size_t)b * L * R2N + r;
  float* py = y + (size_t)b * L * di + d;

  for (int l = 0; l < L; ++l) {
    float dtv = *pdt;
    float xv  = *pxc;
    float zv  = *pz;
    float dx = dtv * xv;
    float yv = 0.f;
#pragma unroll
    for (int n = 0; n < 16; ++n) {
      h[n] = expf(dtv * A[n]) * h[n] + dx * pbc[n];
      yv += h[n] * pbc[16 + n];
    }
    yv = (yv + xv * Dv) * silu_f(zv);
    *py = yv;
    pdt += di; pxc += di; pz += 2 * di; pbc += R2N; py += di;
  }
}

// ===========================================================================
extern "C" void kernel_launch(void* const* d_in, const int* in_sizes, int n_in,
                              void* d_out, int out_size, void* d_ws, size_t ws_size,
                              hipStream_t stream) {
  (void)in_sizes; (void)n_in; (void)out_size; (void)ws_size;
  const int B = 16;

  // ---- workspace carving (fp32) ----
  float* ws = (float*)d_ws;
  size_t off = 0;
  auto alloc = [&](size_t n) { float* p = ws + off; off += (n + 63) & ~(size_t)63; return p; };
  const size_t BIG = (size_t)B * 64 * 16384;      // 16.78M floats
  float* bufx = alloc(BIG);                       // current activation (B,C,L)
  float* bufa = alloc(BIG);                       // conv1 out / xc
  float* bufb = alloc(BIG);                       // conv2 out / dt
  float* bufs = alloc(BIG);                       // skip / LN rows / scan y
  float* bufz = alloc(2 * BIG);                   // in_proj output xz (B*L, 2di)
  float* bufd = alloc((size_t)B * 4096 * 64);     // dbl (B*L, r+2N)
  float* bmean = alloc(512);
  float* bvar  = alloc(512);

  // ---- input walk: setup_inputs() dict insertion order, depth-first ----
  int ii = 0;
  auto next = [&]() { return (const float*)d_in[ii++]; };
  const float* x_in = next();                     // 'x' (16,1,16384)

  struct ResP { const float *w1,*g1,*b1,*w2,*g2,*b2,*wskip; };
  auto read_res = [&](bool skip) {
    ResP p; p.w1 = next(); p.g1 = next(); p.b1 = next();
    p.w2 = next(); p.g2 = next(); p.b2 = next();
    p.wskip = skip ? next() : nullptr; return p;
  };
  struct MamP { const float *ln_g,*ln_b,*in_proj,*conv_w,*conv_b,*x_proj,
                            *dt_w,*dt_b,*A_log,*D,*out_proj; };
  auto read_mamba = [&]() {
    MamP m; m.ln_g = next(); m.ln_b = next(); m.in_proj = next();
    m.conv_w = next(); m.conv_b = next(); m.x_proj = next();
    m.dt_w = next(); m.dt_b = next(); m.A_log = next(); m.D = next();
    m.out_proj = next(); return m;
  };

  ResP stem = read_res(true);
  ResP blk[4][2];
  MamP mam[2];
  for (int s = 0; s < 4; ++s) {
    blk[s][0] = read_res(true);
    blk[s][1] = read_res(false);
    if (s & 1) mam[s >> 1] = read_mamba();
  }

  // ---- launch helpers ----
  auto conv = [&](const float* X, const float* W, float* O,
                  int Cin, int Lin, int Cout, int Lout, int Kw, int pad, int stride) {
    dim3 g((unsigned)(((size_t)B * Lout) / NTILE), (unsigned)(Cout / MTILE));
    conv1d_wmma<<<g, 256, 0, stream>>>(X, W, O, Cin, Lin, Cout, Lout, Kw, pad, stride);
  };
  auto stats = [&](const float* X, int C, int L) {
    bn_stats<<<C, 256, 0, stream>>>(X, bmean, bvar, B, C, L);
  };
  auto bnact = [&](const float* in, const float* res, const float* g, const float* bb,
                   float* out, int C, int L) {
    size_t tot = (size_t)B * C * L;
    bn_act<<<(unsigned)((tot + 255) / 256), 256, 0, stream>>>(
        in, res, bmean, bvar, g, bb, out, C, L, tot);
  };
  auto gemm = [&](const float* act, const float* W, float* O, int Nr, int M, int K,
                  int lda, int ldo, const float* bias, int actmode, int layout, int Lseq) {
    dim3 g((unsigned)(Nr / NTILE), (unsigned)((M + MTILE - 1) / MTILE));
    gemm_wmma<<<g, 256, 0, stream>>>(act, W, O, Nr, M, K, lda, ldo,
                                     bias, actmode, layout, Lseq);
  };

  auto res_block = [&](const float* X, int Cin, int Lin, const ResP& p,
                       int Cout, int stride, float* out) -> int {
    int Lout = Lin / stride;
    conv(X, p.w1, bufa, Cin, Lin, Cout, Lout, 5, 2, stride);   // conv1
    stats(bufa, Cout, Lout);
    bnact(bufa, nullptr, p.g1, p.b1, bufa, Cout, Lout);        // lrelu(bn(.))
    conv(bufa, p.w2, bufb, Cout, Lout, Cout, Lout, 5, 2, 1);   // conv2
    stats(bufb, Cout, Lout);
    const float* res = X;                                      // identity skip
    if (p.wskip) {
      conv(X, p.wskip, bufs, Cin, Lin, Cout, Lout, 1, 0, stride);
      res = bufs;
    }
    bnact(bufb, res, p.g2, p.b2, out, Cout, Lout);             // lrelu(bn+res)
    return Lout;
  };

  auto run_mamba = [&](const float* X, int C, int L, const MamP& m, float* out) {
    const int di = 2 * C;
    const int r = (C + 15) / 16;
    const int R2N = r + 32;
    const int Nr = B * L;
    ln_rows<<<Nr, 128, 0, stream>>>(X, m.ln_g, m.ln_b, bufs, C, L);         // (Nr,C)
    gemm(bufs, m.in_proj, bufz, Nr, 2 * di, C, C, 2 * di, nullptr, 0, 0, 0); // xz
    {
      size_t tot = (size_t)Nr * di;
      dwconv_silu<<<(unsigned)((tot + 255) / 256), 256, 0, stream>>>(
          bufz, m.conv_w, m.conv_b, bufa, L, di, tot);                       // xc
    }
    gemm(bufa, m.x_proj, bufd, Nr, R2N, di, di, R2N, nullptr, 0, 0, 0);      // dbl
    gemm(bufd, m.dt_w, bufb, Nr, di, r, R2N, di, m.dt_b, 1, 0, 0);           // dt
    {
      int th = B * di;
      mamba_scan<<<(th + 255) / 256, 256, 0, stream>>>(
          bufb, bufa, bufz, bufd, m.A_log, m.D, bufs, th, L, di, r);         // y
    }
    gemm(bufs, m.out_proj, out, Nr, C, di, di, 0, nullptr, 0, 1, L);         // (B,C,L)
  };

  // ---- forward ----
  const int FE[4] = {64, 128, 256, 512};
  int C = 64, L = 16384;
  res_block(x_in, 1, 16384, stem, 64, 1, bufx);
  for (int s = 0; s < 4; ++s) {
    int Cout = FE[s];
    L = res_block(bufx, C, L, blk[s][0], Cout, 2, bufx);
    C = Cout;
    res_block(bufx, C, L, blk[s][1], Cout, 1, bufx);
    if (s & 1) {
      float* tgt = (s == 3) ? (float*)d_out : bufx;  // final mamba -> d_out
      run_mamba(bufx, C, L, mam[s >> 1], tgt);
    }
  }
}
